// HashingMemory_51290499448944
// MI455X (gfx1250) — matrix operand order repaired and checked by reading, not compiled, not run on hardware
//
#include <hip/hip_runtime.h>

typedef __attribute__((ext_vector_type(16))) __bf16 v16bf;
typedef __attribute__((ext_vector_type(8)))  __bf16 v8bf;
typedef __attribute__((ext_vector_type(8)))  float  v8f;

#define HEADS   4
#define K_DIM   512
#define HALF_D  256
#define N_KEYS  512
#define KNN     32
#define V_DIM   512
#define IN_DIM  1024
#define BS      2048

#define NEG_INF (-1.0e30f)

// ---------------------------------------------------------------------------
// Prep kernels: f32 -> bf16 convert / transpose
// ---------------------------------------------------------------------------
__global__ void cvt_f32_bf16(const float* __restrict__ in, __bf16* __restrict__ out, int n) {
  int i = blockIdx.x * blockDim.x + threadIdx.x;
  if (i < n) out[i] = (__bf16)in[i];
}

// in: rows x cols (row-major) -> out: cols x rows (row-major), bf16
__global__ void transpose_to_bf16(const float* __restrict__ in, __bf16* __restrict__ out,
                                  int rows, int cols) {
  int i = blockIdx.x * blockDim.x + threadIdx.x;
  if (i < rows * cols) {
    int r = i / cols, c = i % cols;
    out[c * rows + r] = (__bf16)in[i];
  }
}

// ---------------------------------------------------------------------------
// Generic bf16 WMMA GEMM:  C[M,N] = A[M,K] * Bt[N,K]^T  (+bias, optional)
// A, Bt row-major K-contiguous.  Block tile 64(M) x 256(N), BK=32.
// 8 waves (2x4); each wave computes a 32x64 tile = 2x4 WMMA 16x16x32 bf16
// (8 v_wmma per k-step).  Double-buffered LDS, software-pipelined global
// loads: next tile's loads are in flight while this tile's WMMAs execute.
// write_bf16=1: C is bf16, bias added.  write_bf16=0: C is f32, no bias.
// gridDim.z batches with element offsets abatch/bbatch/cbatch.
// ---------------------------------------------------------------------------
__launch_bounds__(256)
__global__ void wmma_gemm_bf16(const __bf16* __restrict__ A, const __bf16* __restrict__ Bt,
                               const float* __restrict__ bias, void* __restrict__ Cout,
                               int K, int lda, int ldb, int ldc,
                               long abatch, long bbatch, long cbatch, int write_bf16)
{
  __shared__ __bf16 As[2][64][40];     // padded rows (80B stride) for bank spread
  __shared__ __bf16 Bs[2][256][40];

  const int tid  = threadIdx.x;
  const int lane = tid & 31;
  const int wv   = tid >> 5;       // 0..7
  const int wm   = wv >> 2;        // 0..1  (M direction)
  const int wn   = wv & 3;         // 0..3  (N direction)
  const int z    = blockIdx.z;

  const __bf16* Ab = A  + (long)z * abatch + (long)(blockIdx.y * 64)  * lda;
  const __bf16* Bb = Bt + (long)z * bbatch + (long)(blockIdx.x * 256) * ldb;

  v8f acc[2][4] = {};

  const int ar = tid >> 2, ac = (tid & 3) * 8;  // A tile: 64x32, 8 elems/thread
  const int br = tid;                           // B tile: 256x32, one row/thread

  const int hl  = lane >> 4;   // lane half
  const int l15 = lane & 15;
  const int nk  = K >> 5;

  // ---- prologue: tile 0 -> LDS buffer 0 ---------------------------------
  v8bf areg, breg0, breg1, breg2, breg3;
  areg  = *(const v8bf*)(Ab + (long)ar * lda + ac);
  breg0 = *(const v8bf*)(Bb + (long)br * ldb + 0);
  breg1 = *(const v8bf*)(Bb + (long)br * ldb + 8);
  breg2 = *(const v8bf*)(Bb + (long)br * ldb + 16);
  breg3 = *(const v8bf*)(Bb + (long)br * ldb + 24);
  *(v8bf*)&As[0][ar][ac] = areg;
  *(v8bf*)&Bs[0][br][0]  = breg0;
  *(v8bf*)&Bs[0][br][8]  = breg1;
  *(v8bf*)&Bs[0][br][16] = breg2;
  *(v8bf*)&Bs[0][br][24] = breg3;
  __syncthreads();

  for (int it = 0; it < nk; ++it) {
    const int  buf     = it & 1;
    const bool hasNext = (it + 1) < nk;

    // issue next tile's global loads; they complete during the WMMAs below
    if (hasNext) {
      const int kt = (it + 1) << 5;
      areg  = *(const v8bf*)(Ab + (long)ar * lda + kt + ac);
      breg0 = *(const v8bf*)(Bb + (long)br * ldb + kt + 0);
      breg1 = *(const v8bf*)(Bb + (long)br * ldb + kt + 8);
      breg2 = *(const v8bf*)(Bb + (long)br * ldb + kt + 16);
      breg3 = *(const v8bf*)(Bb + (long)br * ldb + kt + 24);
    }

    v16bf afrag[2], bfrag[4];
#pragma unroll
    for (int t = 0; t < 2; ++t) {
      // A 16x32 bf16 frag: lane<16 -> K {0..7,16..23}; lane>=16 -> K {8..15,24..31}
      const __bf16* pa = &As[buf][wm * 32 + t * 16 + l15][hl * 8];
      v8bf alo = *(const v8bf*)pa;
      v8bf ahi = *(const v8bf*)(pa + 16);
      afrag[t] = __builtin_shufflevector(alo, ahi, 0,1,2,3,4,5,6,7,8,9,10,11,12,13,14,15);
    }
#pragma unroll
    for (int t = 0; t < 4; ++t) {
      // B 32x16 bf16 frag: lane = N column; lane<16 -> K {0..7,16..23}; lane>=16 -> K {8..15,24..31}
      const __bf16* pb = &Bs[buf][wn * 64 + t * 16 + l15][hl * 8];
      v8bf blo = *(const v8bf*)pb;
      v8bf bhi = *(const v8bf*)(pb + 16);
      bfrag[t] = __builtin_shufflevector(blo, bhi, 0,1,2,3,4,5,6,7,8,9,10,11,12,13,14,15);
    }

#pragma unroll
    for (int ti = 0; ti < 2; ++ti)
#pragma unroll
      for (int tj = 0; tj < 4; ++tj)
        acc[ti][tj] = __builtin_amdgcn_wmma_f32_16x16x32_bf16(
            false, afrag[ti], false, bfrag[tj], (short)0, acc[ti][tj], false, false);

    // stash next tile in the other buffer (current readers untouched)
    if (hasNext) {
      const int nb = buf ^ 1;
      *(v8bf*)&As[nb][ar][ac] = areg;
      *(v8bf*)&Bs[nb][br][0]  = breg0;
      *(v8bf*)&Bs[nb][br][8]  = breg1;
      *(v8bf*)&Bs[nb][br][16] = breg2;
      *(v8bf*)&Bs[nb][br][24] = breg3;
    }
    __syncthreads();
  }

  // Epilogue. C layout: lane<16 -> M=v, lane>=16 -> M=8+v; N = lane&15.
  const long crow0 = (long)blockIdx.y * 64 + wm * 32;
  const long ccol0 = (long)blockIdx.x * 256 + wn * 64;
#pragma unroll
  for (int ti = 0; ti < 2; ++ti) {
#pragma unroll
    for (int tj = 0; tj < 4; ++tj) {
      const long col = ccol0 + tj * 16 + l15;
      float bv = write_bf16 ? bias[col] : 0.0f;
#pragma unroll
      for (int v = 0; v < 8; ++v) {
        const long row = crow0 + ti * 16 + hl * 8 + v;
        float val = acc[ti][tj][v] + bv;
        if (write_bf16)
          ((__bf16*)Cout)[(long)z * cbatch + row * ldc + col] = (__bf16)val;
        else
          ((float*)Cout)[(long)z * cbatch + row * ldc + col] = val;
      }
    }
  }
}

// ---------------------------------------------------------------------------
// Wave-level top-32 of 512 by iterative argmax extraction.
// Lane l holds elements {j*32 + l}. Results (desc order) -> LDS via lane r.
// ---------------------------------------------------------------------------
__device__ __forceinline__ void wave_topk32(const float* __restrict__ s, int lane,
                                            float* __restrict__ ov, int* __restrict__ oi)
{
  float v[16];
#pragma unroll
  for (int j = 0; j < 16; ++j) v[j] = s[j * 32 + lane];
  for (int r = 0; r < 32; ++r) {
    float bv = v[0]; int bj = 0;
#pragma unroll
    for (int j = 1; j < 16; ++j) { if (v[j] > bv) { bv = v[j]; bj = j; } }
    int bidx = bj * 32 + lane;
#pragma unroll
    for (int off = 16; off > 0; off >>= 1) {
      float xv = __shfl_xor(bv, off, 32);
      int   xi = __shfl_xor(bidx, off, 32);
      if (xv > bv || (xv == bv && xi < bidx)) { bv = xv; bidx = xi; }
    }
    if ((bidx & 31) == lane) {          // winner lane masks its element
      int mj = bidx >> 5;
#pragma unroll
      for (int j = 0; j < 16; ++j) if (j == mj) v[j] = NEG_INF;
    }
    if (lane == r) { ov[r] = bv; oi[r] = bidx; }
  }
}

// One wave per (b,h): top32(s1), top32(s2), top32 of 1024 pair-sums, softmax.
__launch_bounds__(256)
__global__ void topk_combine(const float* __restrict__ S, float* __restrict__ Wout,
                             int* __restrict__ Iout)
{
  __shared__ float sv1[8][32]; __shared__ int si1[8][32];
  __shared__ float sv2[8][32]; __shared__ int si2[8][32];
  const int tid = threadIdx.x, lane = tid & 31, wv = tid >> 5;
  const int pair = blockIdx.x * 8 + wv;      // 0..8191
  const int b = pair >> 2, h = pair & 3;
  const float* s1 = S + (long)b * (HEADS * 2 * N_KEYS) + h * (2 * N_KEYS);
  const float* s2 = s1 + N_KEYS;

  wave_topk32(s1, lane, sv1[wv], si1[wv]);
  wave_topk32(s2, lane, sv2[wv], si2[wv]);
  __syncthreads();

  // candidate grid: i = lane (s1 rank), j = 0..31 (s2 rank)
  float cand[32];
  {
    float a = sv1[wv][lane];
#pragma unroll
    for (int j = 0; j < 32; ++j) cand[j] = a + sv2[wv][j];
  }
  float selv = NEG_INF; int seli = 0;
  for (int r = 0; r < 32; ++r) {
    float bv = cand[0]; int bj = 0;
#pragma unroll
    for (int j = 1; j < 32; ++j) { if (cand[j] > bv) { bv = cand[j]; bj = j; } }
    int bidx = lane * 32 + bj;
#pragma unroll
    for (int off = 16; off > 0; off >>= 1) {
      float xv = __shfl_xor(bv, off, 32);
      int   xi = __shfl_xor(bidx, off, 32);
      if (xv > bv || (xv == bv && xi < bidx)) { bv = xv; bidx = xi; }
    }
    if ((bidx >> 5) == lane) {
      int mj = bidx & 31;
#pragma unroll
      for (int j = 0; j < 32; ++j) if (j == mj) cand[j] = NEG_INF;
    }
    if (lane == r) { selv = bv; seli = bidx; }
  }

  float vmax = __shfl(selv, 0, 32);          // first extracted = global max
  float e = __expf(selv - vmax);
  float sum = e;
#pragma unroll
  for (int off = 16; off > 0; off >>= 1) sum += __shfl_xor(sum, off, 32);
  float wgt = e / sum;

  int i1 = si1[wv][seli >> 5];
  int i2 = si2[wv][seli & 31];
  int o = b * (HEADS * KNN) + h * KNN + lane;
  Wout[o] = wgt;
  Iout[o] = i1 * N_KEYS + i2;
}

// ---------------------------------------------------------------------------
// out[b,:] = sum_k w[b,k] * values[idx[b,k], :]   (HBM-bound: 512MB streamed)
// global_prefetch_b8 lookahead hides the data-dependent gather latency.
// ---------------------------------------------------------------------------
__launch_bounds__(128)
__global__ void gather_weighted(const float* __restrict__ values, const float* __restrict__ W,
                                const int* __restrict__ I, float* __restrict__ out)
{
  __shared__ float w[128];
  __shared__ int  id[128];
  const int b = blockIdx.x, tid = threadIdx.x;
  w[tid]  = W[b * 128 + tid];
  id[tid] = I[b * 128 + tid];
  __syncthreads();

  // warm the pipe: prefetch the first few rows
#pragma unroll
  for (int k = 0; k < 8; ++k)
    __builtin_prefetch(values + (long)id[k] * V_DIM + tid * 4, 0, 0);

  float4 acc = make_float4(0.f, 0.f, 0.f, 0.f);
#pragma unroll 4
  for (int k = 0; k < 128; ++k) {
    if (k + 8 < 128)
      __builtin_prefetch(values + (long)id[k + 8] * V_DIM + tid * 4, 0, 0);
    const float4* row = (const float4*)(values + (long)id[k] * V_DIM);
    float4 v = row[tid];
    float ww = w[k];
    acc.x += ww * v.x; acc.y += ww * v.y; acc.z += ww * v.z; acc.w += ww * v.w;
  }
  ((float4*)(out + (long)b * V_DIM))[tid] = acc;
}

// ---------------------------------------------------------------------------
extern "C" void kernel_launch(void* const* d_in, const int* in_sizes, int n_in,
                              void* d_out, int out_size, void* d_ws, size_t ws_size,
                              hipStream_t stream) {
  const float* x      = (const float*)d_in[0];   // 2048 x 1024
  const float* Wq     = (const float*)d_in[1];   // 1024 x 2048
  const float* bq     = (const float*)d_in[2];   // 2048
  const float* keys   = (const float*)d_in[3];   // 4 x 2 x 512 x 256
  const float* values = (const float*)d_in[4];   // 262144 x 512
  float* out = (float*)d_out;                    // 2048 x 512

  char* ws = (char*)d_ws;
  size_t off = 0;
  auto alloc = [&](size_t bytes) -> void* {
    void* p = ws + off;
    off += (bytes + 255) & ~(size_t)255;
    return p;
  };
  __bf16* xb  = (__bf16*)alloc((size_t)BS * IN_DIM * 2);                 // 4 MB
  __bf16* wqt = (__bf16*)alloc((size_t)(HEADS * K_DIM) * IN_DIM * 2);    // 4 MB (N x K)
  __bf16* kb  = (__bf16*)alloc((size_t)HEADS * 2 * N_KEYS * HALF_D * 2); // 2 MB
  __bf16* qb  = (__bf16*)alloc((size_t)BS * HEADS * K_DIM * 2);          // 8 MB
  float*  S   = (float*)alloc((size_t)BS * HEADS * 2 * N_KEYS * 4);      // 32 MB
  float*  Wv  = (float*)alloc((size_t)BS * HEADS * KNN * 4);             // 1 MB
  int*    Iv  = (int*)alloc((size_t)BS * HEADS * KNN * 4);               // 1 MB
  (void)ws_size; (void)in_sizes; (void)n_in; (void)out_size;

  // ---- prep: bf16 conversions -------------------------------------------
  {
    int n = BS * IN_DIM;
    cvt_f32_bf16<<<(n + 255) / 256, 256, 0, stream>>>(x, xb, n);
  }
  {
    int n = IN_DIM * HEADS * K_DIM;
    transpose_to_bf16<<<(n + 255) / 256, 256, 0, stream>>>(Wq, wqt, IN_DIM, HEADS * K_DIM);
  }
  {
    int n = HEADS * 2 * N_KEYS * HALF_D;
    cvt_f32_bf16<<<(n + 255) / 256, 256, 0, stream>>>(keys, kb, n);
  }

  // ---- GEMM1: q = x @ Wq + bq  (M=2048,N=2048,K=1024), bf16 out ----------
  wmma_gemm_bf16<<<dim3((HEADS * K_DIM) / 256, BS / 64, 1), 256, 0, stream>>>(
      xb, wqt, bq, qb,
      /*K=*/IN_DIM, /*lda=*/IN_DIM, /*ldb=*/IN_DIM, /*ldc=*/HEADS * K_DIM,
      0, 0, 0, /*write_bf16=*/1);

  // ---- GEMM2: S[b,z,n] = q[b, z*256 : z*256+256] . keys[z][n,:]  z=h*2+p --
  wmma_gemm_bf16<<<dim3(N_KEYS / 256, BS / 64, HEADS * 2), 256, 0, stream>>>(
      qb, kb, nullptr, S,
      /*K=*/HALF_D, /*lda=*/HEADS * K_DIM, /*ldb=*/HALF_D, /*ldc=*/HEADS * 2 * N_KEYS,
      /*abatch=*/HALF_D, /*bbatch=*/(long)N_KEYS * HALF_D, /*cbatch=*/N_KEYS,
      /*write_bf16=*/0);

  // ---- top-k + combine + softmax ----------------------------------------
  topk_combine<<<(BS * HEADS) / 8, 256, 0, stream>>>(S, Wv, Iv);

  // ---- weighted gather (HBM-bound) --------------------------------------
  gather_weighted<<<BS, 128, 0, stream>>>(values, Wv, Iv, out);
}
